// NSE_2070174237287
// MI455X (gfx1250) — matrix operand order repaired and checked by reading, not compiled, not run on hardware
//
#include <hip/hip_runtime.h>
#include <hip/hip_bf16.h>

// ---------------------------------------------------------------------------
// NSE recurrent network on MI455X (gfx1250, wave32, WMMA).
// Strategy: prep kernel packs weights/embeddings into WMMA-native f16 tile
// layouts; a persistent 16-WG kernel runs all 64 timesteps with an atomic
// grid barrier, doing every GEMM via v_wmma_f32_16x16x32_f16.
// Padded dims: DW 300->320, 2DW 600->640, DC 200->224(K)/208(N), NR 19->32.
// ---------------------------------------------------------------------------

typedef __attribute__((ext_vector_type(16))) _Float16 v16h;
typedef __attribute__((ext_vector_type(8)))  float    v8f;

#define NWG 16
#define NTH 256

struct NseParams {
  // inputs
  const int*   x;
  const float* emb;
  const float *Wihr, *Whhr, *bihr, *bhhr;
  const float *Wc,   *bc;
  const float *Wihw, *Whhw, *bihw, *bhhw;
  const float *W1, *b1, *W2, *b2;
  // workspace (fp32)
  int *bar, *sense;
  float *Mmem;                       // [64*64][320] padded memory
  float *gates;                      // [64][1200]
  float *h_r, *c_r, *h_w, *c_w;      // [64][320] padded states
  float *sbuf, *zbuf;                // [64][64] scores / softmax
  float *cin, *ct;                   // [64][640]
  float *abuf;                       // [64][224]
  float *ybuf;                       // [64][32]
  // workspace (f16, WMMA tile layouts)
  _Float16 *XA;                      // [64 steps][4mt][10kt][32lane][16]
  _Float16 *hrA, *hwA;               // [4][10][32][16]
  _Float16 *cinA, *ctA;              // [4][20][32][16]
  _Float16 *aA;                      // [4][7][32][16]
  _Float16 *BihR, *BhhR, *Bc_, *BihW, *BhhW, *BW1, *BW2;
  float* out;
};

// ---- WMMA helpers ---------------------------------------------------------

// One K-strip of 16x16 output tile: acc += A(mt, kt..) x B(nt, kt..)
__device__ inline void gemmK(const _Float16* A, const _Float16* Bw, int KT,
                             v8f& acc, int mt, int nt, int lane) {
  for (int kt = 0; kt < KT; ++kt) {
    v16h a = *(const v16h*)(A + ((((size_t)mt * KT + kt) * 32 + lane) << 4));
    v16h b = *(const v16h*)(Bw + ((((size_t)nt * KT + kt) * 32 + lane) << 4));
    acc = __builtin_amdgcn_wmma_f32_16x16x32_f16(
        /*neg_a=*/false, a, /*neg_b=*/false, b,
        /*c_mod=*/(short)0, acc, /*reuse_a=*/false, /*reuse_b=*/false);
  }
}

// C/D layout: VGPR i -> row mt*16 + i + (lane>=16 ? 8 : 0), col nt*16 + lane%16
__device__ inline void store_tile(float* out, int ldo, v8f acc, int mt, int nt, int lane) {
  int n  = nt * 16 + (lane & 15);
  int m0 = mt * 16 + ((lane >> 4) << 3);
#pragma unroll
  for (int i = 0; i < 8; ++i) out[(size_t)(m0 + i) * ldo + n] = acc[i];
}

// Pack fp32 [M][Kp] row-major into A-matrix 16x32 f16 wave layout:
// lane<16: elems 0..7 -> K kt*32+0..7, elems 8..15 -> K kt*32+16..23
// lane>=16: +8 on each chunk.  Row = mt*16 + lane%16.
__device__ inline void pack_A(const float* src, int Kp, _Float16* dst,
                              int MT, int KT, int tid, int nth) {
  const int items = MT * KT * 32;
  for (int it = tid; it < items; it += nth) {
    int lane = it & 31;
    int q = it >> 5;
    int kt = q % KT;
    int mt = q / KT;
    const float* s = src + (size_t)(mt * 16 + (lane & 15)) * Kp
                         + kt * 32 + ((lane >> 4) << 3);
    _Float16* d = dst + ((size_t)it << 4);
#pragma unroll
    for (int e = 0; e < 8; ++e) d[e] = (_Float16)s[e];
#pragma unroll
    for (int e = 0; e < 8; ++e) d[8 + e] = (_Float16)s[16 + e];
  }
}

// Pack weight W [Nrows][Ktot] (row-major, out x in) into B-matrix 32x16 layout:
// lane = column n (lane%16), K = kt*32 + (lane>=16?8:0) + e + (e>=8?8:0).
// Padded-K mapping: k<seg0 -> k ; segPad<=k<segPad+seg1 -> seg0+(k-segPad); else 0.
__device__ inline void pack_Bm(const float* W, int Nrows, int seg0, int seg1, int segPad,
                               _Float16* dst, int NT, int KT, int tid, int nth) {
  const int Ktot = seg0 + seg1;
  const int items = NT * KT * 32;
  for (int it = tid; it < items; it += nth) {
    int lane = it & 31;
    int q = it >> 5;
    int kt = q % KT;
    int nt = q / KT;
    int n  = nt * 16 + (lane & 15);
    int kb = kt * 32 + ((lane >> 4) << 3);
    _Float16* d = dst + ((size_t)it << 4);
#pragma unroll
    for (int e = 0; e < 16; ++e) {
      int k = kb + e + ((e >> 3) << 3);
      float v = 0.f;
      if (n < Nrows) {
        int ko = -1;
        if (k < seg0) ko = k;
        else if (seg1 && k >= segPad && k < segPad + seg1) ko = seg0 + (k - segPad);
        if (ko >= 0) v = W[(size_t)n * Ktot + ko];
      }
      d[e] = (_Float16)v;
    }
  }
}

// ---- pointwise helpers ----------------------------------------------------

__device__ inline float sigm(float x) { return 1.f / (1.f + __expf(-x)); }

__device__ inline void lstm_point(const float* gates, const float* bih, const float* bhh,
                                  float* h, float* c, int tid, int nth) {
  for (int idx = tid; idx < 64 * 300; idx += nth) {
    int b = idx / 300, d = idx % 300;
    const float* g = gates + b * 1200;
    float ig = g[d]       + bih[d]       + bhh[d];
    float fg = g[300 + d] + bih[300 + d] + bhh[300 + d];
    float gg = g[600 + d] + bih[600 + d] + bhh[600 + d];
    float og = g[900 + d] + bih[900 + d] + bhh[900 + d];
    float cn = sigm(fg) * c[b * 320 + d] + sigm(ig) * tanhf(gg);
    c[b * 320 + d] = cn;
    h[b * 320 + d] = sigm(og) * tanhf(cn);
  }
}

// Sense-reversal grid barrier across NWG resident workgroups.
__device__ inline void gbar(int* cnt, int* sense_g, int& sense) {
  int s = sense ^ 1;
  __syncthreads();
  if (threadIdx.x == 0) {
    if (__hip_atomic_fetch_add(cnt, 1, __ATOMIC_ACQ_REL, __HIP_MEMORY_SCOPE_AGENT) == NWG - 1) {
      __hip_atomic_store(cnt, 0, __ATOMIC_RELAXED, __HIP_MEMORY_SCOPE_AGENT);
      __hip_atomic_store(sense_g, s, __ATOMIC_RELEASE, __HIP_MEMORY_SCOPE_AGENT);
    } else {
      while (__hip_atomic_load(sense_g, __ATOMIC_ACQUIRE, __HIP_MEMORY_SCOPE_AGENT) != s)
        __builtin_amdgcn_s_sleep(1);
    }
  }
  __syncthreads();
  sense = s;
}

// ---------------------------------------------------------------------------
// Kernel 1: prep — pack weights, gather embeddings, zero recurrent state.
// ---------------------------------------------------------------------------
__global__ __launch_bounds__(NTH) void nse_prep(NseParams p) {
  const int tid = blockIdx.x * blockDim.x + threadIdx.x;
  const int nth = gridDim.x * blockDim.x;
  if (tid == 0) { *p.bar = 0; *p.sense = 0; }

  // M0 = emb[x] padded to [64*64][320]
  for (int idx = tid; idx < 64 * 64 * 320; idx += nth) {
    int bn = idx / 320, d = idx % 320;
    p.Mmem[idx] = (d < 300) ? p.emb[(size_t)p.x[bn] * 300 + d] : 0.f;
  }
  // XA[t] = A-layout f16 of x_t = emb[x[:, t]] (constant scan input)
  for (int it = tid; it < 64 * 4 * 10 * 32; it += nth) {
    int lane = it & 31;
    int q = it >> 5;
    int kt = q % 10; q /= 10;
    int mt = q % 4;
    int t  = q / 4;
    int b  = mt * 16 + (lane & 15);
    int kb = kt * 32 + ((lane >> 4) << 3);
    const float* e = p.emb + (size_t)p.x[b * 64 + t] * 300;
    _Float16* d = p.XA + ((size_t)it << 4);
#pragma unroll
    for (int i = 0; i < 8; ++i) { int k = kb + i;      d[i]     = (k < 300) ? (_Float16)e[k] : (_Float16)0.f; }
#pragma unroll
    for (int i = 0; i < 8; ++i) { int k = kb + 16 + i; d[8 + i] = (k < 300) ? (_Float16)e[k] : (_Float16)0.f; }
  }
  // zero recurrent state + padded staging buffers
  for (int idx = tid; idx < 20480; idx += nth) {
    p.h_r[idx] = 0.f; p.c_r[idx] = 0.f; p.h_w[idx] = 0.f; p.c_w[idx] = 0.f;
    p.hrA[idx] = (_Float16)0.f; p.hwA[idx] = (_Float16)0.f;
  }
  for (int idx = tid; idx < 40960; idx += nth) { p.cin[idx] = 0.f; p.ct[idx] = 0.f; }
  for (int idx = tid; idx < 14336; idx += nth) p.abuf[idx] = 0.f;

  // weights -> B-matrix layouts (K padded; compose input is two 320-padded segs)
  pack_Bm(p.Wihr, 1200, 300,   0,   0, p.BihR, 75, 10, tid, nth);
  pack_Bm(p.Whhr, 1200, 300,   0,   0, p.BhhR, 75, 10, tid, nth);
  pack_Bm(p.Wc,    600, 300, 300, 320, p.Bc_,  38, 20, tid, nth);
  pack_Bm(p.Wihw, 1200, 600,   0,   0, p.BihW, 75, 20, tid, nth);
  pack_Bm(p.Whhw, 1200, 300,   0,   0, p.BhhW, 75, 10, tid, nth);
  pack_Bm(p.W1,    200, 300,   0,   0, p.BW1,  13, 10, tid, nth);
  pack_Bm(p.W2,     19, 200,   0,   0, p.BW2,   2,  7, tid, nth);
}

// ---------------------------------------------------------------------------
// Kernel 2: persistent recurrence (16 WGs x 256 thr = 128 wave32 waves).
// ---------------------------------------------------------------------------
__global__ __launch_bounds__(NTH) void nse_main(NseParams p) {
  const int tid    = blockIdx.x * NTH + threadIdx.x;
  const int nth    = NWG * NTH;
  const int lane   = threadIdx.x & 31;
  const int wave   = tid >> 5;
  const int nwaves = nth >> 5;
  int sense = 0;

  for (int t = 0; t < 64; ++t) {
    const _Float16* xA = p.XA + (size_t)t * (4 * 10 * 32 * 16);

    // P1: read-LSTM gates = x_t @ Wih_r^T + h_r @ Whh_r^T  (4x75 tiles)
    for (int tile = wave; tile < 300; tile += nwaves) {
      int mt = tile / 75, nt = tile % 75;
      v8f acc = {};
      gemmK(xA,    p.BihR, 10, acc, mt, nt, lane);
      gemmK(p.hrA, p.BhhR, 10, acc, mt, nt, lane);
      store_tile(p.gates, 1200, acc, mt, nt, lane);
    }
    gbar(p.bar, p.sense, sense);

    lstm_point(p.gates, p.bihr, p.bhhr, p.h_r, p.c_r, tid, nth);   // o_t = h_r
    gbar(p.bar, p.sense, sense);

    // P2a: attention scores s[b][n] = <M[b][n], o_t[b]>; also repack h_r for next step
    for (int idx = tid; idx < 64 * 64; idx += nth) {
      int b = idx >> 6;
      const float* Mr = p.Mmem + (size_t)idx * 320;
      const float* h  = p.h_r + b * 320;
      float s = 0.f;
      for (int d = 0; d < 300; ++d) s += Mr[d] * h[d];
      p.sbuf[idx] = s;
    }
    pack_A(p.h_r, 320, p.hrA, 4, 10, tid, nth);
    gbar(p.bar, p.sense, sense);

    // P2b: softmax over n (one thread per batch row)
    if (tid < 64) {
      const float* s = p.sbuf + tid * 64;
      float* z = p.zbuf + tid * 64;
      float mx = -1e30f;
      for (int n = 0; n < 64; ++n) mx = fmaxf(mx, s[n]);
      float sum = 0.f;
      for (int n = 0; n < 64; ++n) { float e = __expf(s[n] - mx); z[n] = e; sum += e; }
      float inv = 1.f / sum;
      for (int n = 0; n < 64; ++n) z[n] *= inv;
    }
    gbar(p.bar, p.sense, sense);

    // P2c: m_t = z^T M ; build concat input [o_t | m_t] (two 320-padded segments)
    for (int idx = tid; idx < 64 * 300; idx += nth) {
      int b = idx / 300, d = idx % 300;
      p.cin[b * 640 + d] = p.h_r[b * 320 + d];
      const float* z  = p.zbuf + b * 64;
      const float* Mb = p.Mmem + (size_t)b * 64 * 320 + d;
      float m = 0.f;
      for (int n = 0; n < 64; ++n) m += z[n] * Mb[n * 320];
      p.cin[b * 640 + 320 + d] = m;
    }
    gbar(p.bar, p.sense, sense);

    pack_A(p.cin, 640, p.cinA, 4, 20, tid, nth);
    gbar(p.bar, p.sense, sense);

    // P3a: compose c_t = concat @ Wc^T + bc  (4x38 tiles, bias fused in store)
    for (int tile = wave; tile < 4 * 38; tile += nwaves) {
      int mt = tile / 38, nt = tile % 38;
      v8f acc = {};
      gemmK(p.cinA, p.Bc_, 20, acc, mt, nt, lane);
      int n  = nt * 16 + (lane & 15);
      int m0 = mt * 16 + ((lane >> 4) << 3);
      float bias = (n < 600) ? p.bc[n] : 0.f;
#pragma unroll
      for (int i = 0; i < 8; ++i) p.ct[(size_t)(m0 + i) * 640 + n] = acc[i] + bias;
    }
    gbar(p.bar, p.sense, sense);

    pack_A(p.ct, 640, p.ctA, 4, 20, tid, nth);
    gbar(p.bar, p.sense, sense);

    // P3c: write-LSTM gates = c_t @ Wih_w^T + h_w @ Whh_w^T
    for (int tile = wave; tile < 300; tile += nwaves) {
      int mt = tile / 75, nt = tile % 75;
      v8f acc = {};
      gemmK(p.ctA, p.BihW, 20, acc, mt, nt, lane);
      gemmK(p.hwA, p.BhhW, 10, acc, mt, nt, lane);
      store_tile(p.gates, 1200, acc, mt, nt, lane);
    }
    gbar(p.bar, p.sense, sense);

    lstm_point(p.gates, p.bihw, p.bhhw, p.h_w, p.c_w, tid, nth);   // h_t = h_w
    gbar(p.bar, p.sense, sense);

    // P4: M = (1-z) M + z h_t ; repack h_w for next step's recurrent GEMM
    pack_A(p.h_w, 320, p.hwA, 4, 10, tid, nth);
    for (int idx = tid; idx < 64 * 64 * 300; idx += nth) {
      int bn = idx / 300, d = idx % 300;
      float zz = p.zbuf[bn];
      float* Mp = p.Mmem + (size_t)bn * 320 + d;
      *Mp = (1.f - zz) * (*Mp) + zz * p.h_w[(bn >> 6) * 320 + d];
    }
    gbar(p.bar, p.sense, sense);
  }

  // P5: MLP  y = relu(h @ W1^T + b1) @ W2^T + b2   (bias+relu fused in store)
  for (int tile = wave; tile < 4 * 13; tile += nwaves) {
    int mt = tile / 13, nt = tile % 13;
    v8f acc = {};
    gemmK(p.hwA, p.BW1, 10, acc, mt, nt, lane);
    int n  = nt * 16 + (lane & 15);
    int m0 = mt * 16 + ((lane >> 4) << 3);
    float bias = (n < 200) ? p.b1[n] : 0.f;
#pragma unroll
    for (int i = 0; i < 8; ++i)
      p.abuf[(size_t)(m0 + i) * 224 + n] = fmaxf(acc[i] + bias, 0.f);
  }
  gbar(p.bar, p.sense, sense);

  pack_A(p.abuf, 224, p.aA, 4, 7, tid, nth);
  gbar(p.bar, p.sense, sense);

  for (int tile = wave; tile < 8; tile += nwaves) {
    int mt = tile >> 1, nt = tile & 1;
    v8f acc = {};
    gemmK(p.aA, p.BW2, 7, acc, mt, nt, lane);
    int n  = nt * 16 + (lane & 15);
    int m0 = mt * 16 + ((lane >> 4) << 3);
    float bias = (n < 19) ? p.b2[n] : 0.f;
#pragma unroll
    for (int i = 0; i < 8; ++i) p.ybuf[(m0 + i) * 32 + n] = acc[i] + bias;
  }
  gbar(p.bar, p.sense, sense);

  // write outputs: y [64][19] then M [64][64][300]
  for (int idx = tid; idx < 64 * 19; idx += nth)
    p.out[idx] = p.ybuf[(idx / 19) * 32 + (idx % 19)];
  for (int idx = tid; idx < 64 * 64 * 300; idx += nth)
    p.out[1216 + idx] = p.Mmem[(size_t)(idx / 300) * 320 + (idx % 300)];
}

// ---------------------------------------------------------------------------
extern "C" void kernel_launch(void* const* d_in, const int* in_sizes, int n_in,
                              void* d_out, int out_size, void* d_ws, size_t ws_size,
                              hipStream_t stream) {
  (void)in_sizes; (void)n_in; (void)out_size; (void)ws_size;
  NseParams p{};
  p.x    = (const int*)d_in[0];
  p.emb  = (const float*)d_in[5];
  p.Wihr = (const float*)d_in[6];  p.Whhr = (const float*)d_in[7];
  p.bihr = (const float*)d_in[8];  p.bhhr = (const float*)d_in[9];
  p.Wc   = (const float*)d_in[10]; p.bc   = (const float*)d_in[11];
  p.Wihw = (const float*)d_in[12]; p.Whhw = (const float*)d_in[13];
  p.bihw = (const float*)d_in[14]; p.bhhw = (const float*)d_in[15];
  p.W1   = (const float*)d_in[16]; p.b1   = (const float*)d_in[17];
  p.W2   = (const float*)d_in[18]; p.b2   = (const float*)d_in[19];
  p.out  = (float*)d_out;

  char* w = (char*)d_ws;
  size_t off = 0;
  auto carve = [&](size_t bytes) -> void* {
    void* r = w + off;
    off = (off + bytes + 255) & ~(size_t)255;
    return r;
  };
  p.bar   = (int*)carve(8);          p.sense = p.bar + 1;
  p.Mmem  = (float*)carve(64ull * 64 * 320 * 4);
  p.gates = (float*)carve(64ull * 1200 * 4);
  p.h_r   = (float*)carve(20480 * 4);
  p.c_r   = (float*)carve(20480 * 4);
  p.h_w   = (float*)carve(20480 * 4);
  p.c_w   = (float*)carve(20480 * 4);
  p.sbuf  = (float*)carve(4096 * 4);
  p.zbuf  = (float*)carve(4096 * 4);
  p.cin   = (float*)carve(40960 * 4);
  p.ct    = (float*)carve(40960 * 4);
  p.abuf  = (float*)carve(14336 * 4);
  p.ybuf  = (float*)carve(2048 * 4);
  p.XA    = (_Float16*)carve(64ull * 4 * 10 * 32 * 16 * 2);
  p.hrA   = (_Float16*)carve(20480 * 2);
  p.hwA   = (_Float16*)carve(20480 * 2);
  p.cinA  = (_Float16*)carve(40960 * 2);
  p.ctA   = (_Float16*)carve(40960 * 2);
  p.aA    = (_Float16*)carve(14336 * 2);
  p.BihR  = (_Float16*)carve(75ull * 10 * 32 * 16 * 2);
  p.BhhR  = (_Float16*)carve(75ull * 10 * 32 * 16 * 2);
  p.Bc_   = (_Float16*)carve(38ull * 20 * 32 * 16 * 2);
  p.BihW  = (_Float16*)carve(75ull * 20 * 32 * 16 * 2);
  p.BhhW  = (_Float16*)carve(75ull * 10 * 32 * 16 * 2);
  p.BW1   = (_Float16*)carve(13ull * 10 * 32 * 16 * 2);
  p.BW2   = (_Float16*)carve(2ull  *  7 * 32 * 16 * 2);

  nse_prep<<<512, NTH, 0, stream>>>(p);
  nse_main<<<NWG, NTH, 0, stream>>>(p);
}
